// Net_1975684956437
// MI455X (gfx1250) — compile-verified, matrix operand order checked by reading, not run on hardware
//
#include <hip/hip_runtime.h>
#include <hip/hip_bf16.h>
#include <stddef.h>

// ---------------------------------------------------------------------------
// CDNA5 (gfx1250) implementation.
//  - Phase 1: per-atom tanh/g-table kernel + energy reduction (tiny).
//  - Phase 2: per-pair dot products via v_wmma_f32_16x16x32_bf16 (diagonal
//    trick). coeffs_derivs (384 MB, the bandwidth bound) is loaded straight
//    from global in WMMA register order (cacheline-coalesced); the gathered
//    g-rows (B operand) are staged through LDS with async global->LDS loads.
//    bf16 packing uses v_cvt_pk_bf16_f32 (1 VALU / 2 elems, RNE) via inline
//    asm since this toolchain doesn't declare the builtin.
// ---------------------------------------------------------------------------

typedef __attribute__((ext_vector_type(16))) __bf16    v16bf;
typedef __attribute__((ext_vector_type(8)))  float     v8f;
typedef __attribute__((ext_vector_type(8)))  unsigned  v8u;
typedef __attribute__((ext_vector_type(4)))  int       v4i;

#define NDESC 64

#ifdef __has_builtin
# if __has_builtin(__builtin_amdgcn_global_load_async_to_lds_b128) && \
     __has_builtin(__builtin_amdgcn_s_wait_asynccnt)
#  define USE_ASYNC 1
# endif
#endif
#ifndef USE_ASYNC
# define USE_ASYNC 0
#endif

// Pack two f32 -> two bf16 (RNE) in one dword with v_cvt_pk_bf16_f32.
static __device__ __forceinline__ unsigned pkbf(float lo, float hi) {
  unsigned r;
  asm("v_cvt_pk_bf16_f32 %0, %1, %2" : "=v"(r) : "v"(lo), "v"(hi));
  return r;
}

// Build one v16bf WMMA operand half from a row pointer (global or LDS).
//  Element order: e=0..7 -> k = base+0..7 ; e=8..15 -> k = base+16..23
//  where base = koff + cb (cb = 0 for lanes 0-15, 8 for lanes 16-31).
static __device__ __forceinline__ v16bf pack_half(const float* rowp, int off) {
  const float4* r0 = (const float4*)(rowp + off);        // k = off .. off+7
  const float4* r1 = (const float4*)(rowp + off + 16);   // k = off+16 .. +23
  float4 q0 = r0[0], q1 = r0[1];
  float4 q2 = r1[0], q3 = r1[1];
  v8u pk;
  pk[0] = pkbf(q0.x, q0.y);
  pk[1] = pkbf(q0.z, q0.w);
  pk[2] = pkbf(q1.x, q1.y);
  pk[3] = pkbf(q1.z, q1.w);
  pk[4] = pkbf(q2.x, q2.y);
  pk[5] = pkbf(q2.z, q2.w);
  pk[6] = pkbf(q3.x, q3.y);
  pk[7] = pkbf(q3.z, q3.w);
  return __builtin_bit_cast(v16bf, pk);
}

// out[0] = b_last, out[1..] = 0
__global__ void init_out_kernel(float* __restrict__ out,
                                const float* __restrict__ b_last, int n) {
  int i = blockIdx.x * blockDim.x + threadIdx.x;
  if (i < n) out[i] = (i == 0) ? b_last[0] : 0.0f;
}

// Per-atom: g[i] = (1 - tanh^2)*w1*w_last, energy += tanh*w_last / N
__global__ __launch_bounds__(256)
void atom_kernel(const float* __restrict__ coeffs,
                 const float* __restrict__ w1,
                 const float* __restrict__ b1,
                 const float* __restrict__ wl,
                 float* __restrict__ g,
                 float* __restrict__ energy_out,
                 int total, float invN) {
  __shared__ float red[8];
  float part = 0.0f;
  for (int i = blockIdx.x * 256 + threadIdx.x; i < total;
       i += gridDim.x * 256) {
    int d = i & (NDESC - 1);
    float e = tanhf(coeffs[i] * w1[d] + b1[d]);
    float wld = wl[d];
    g[i] = (1.0f - e * e) * w1[d] * wld;
    part += e * wld;
  }
  for (int off = 16; off > 0; off >>= 1) part += __shfl_down(part, off, 32);
  if ((threadIdx.x & 31) == 0) red[threadIdx.x >> 5] = part;
  __syncthreads();
  if (threadIdx.x < 8) {
    float v = red[threadIdx.x];
    for (int off = 4; off > 0; off >>= 1) v += __shfl_down(v, off, 8);
    if (threadIdx.x == 0) atomicAdd(energy_out, v * invN);
  }
}

// Main streaming kernel: one wave32 per 16-pair tile.
// D = A(16x64 cd tile, bf16) x B(64x16 gathered g rows, bf16); diag(D) = dots.
__global__ __launch_bounds__(128)
void pair_wmma_kernel(const float* __restrict__ cd,     // [3, P, 64]
                      const float* __restrict__ g,      // [NA, 64]
                      const int*   __restrict__ central,
                      const int*   __restrict__ neigh,
                      float* __restrict__ outF,         // [3, NA]
                      int P, int NA, int nTiles) {
  __shared__ float lds[4 * 1024];                 // 4 waves * B tile (16x64)
  const int lane = threadIdx.x & 31;
  const int wave = threadIdx.x >> 5;
  const int tile = blockIdx.x * 4 + wave;
  if (tile >= nTiles) return;                     // wave-uniform exit

  float* gT = &lds[wave * 1024];                  // B tile: 16 rows x 64 f32
  const int p0 = tile * 16;

  // Gather g rows for the 16 central atoms into LDS (async DMA path).
  for (int i = 0; i < 8; ++i) {
    int u = i * 32 + lane;                        // float4 index within tile
    int row = u >> 4;
    int c4 = u & 15;
    int cidx = central[p0 + row];
#if USE_ASYNC
    __builtin_amdgcn_global_load_async_to_lds_b128(
        (v4i*)(g + (size_t)cidx * NDESC + c4 * 4),
        (v4i*)(gT + u * 4), 0, 0);
#else
    ((float4*)gT)[u] = ((const float4*)(g + (size_t)cidx * NDESC))[c4];
#endif
  }
#if USE_ASYNC
  __builtin_amdgcn_s_wait_asynccnt(0);
#else
  asm volatile("s_wait_dscnt 0" ::: "memory");
#endif

  // Per-lane WMMA coordinates: A row / B column = lane&15,
  // K-group base cb = 0 (lanes 0-15) or 8 (lanes 16-31).
  const int mn = lane & 15;
  const int cb = (lane >> 4) ? 8 : 0;

  // Pack B operand once (reused for all 3 dims).
  const float* gRow = gT + mn * NDESC;
  v16bf bmat[2];
  bmat[0] = pack_half(gRow, 0 + cb);
  bmat[1] = pack_half(gRow, 32 + cb);

  const bool own = (lane < 8) || (lane >= 24);
  const int  dp  = (lane < 8) ? lane : (lane - 16);   // diagonal pair index
  const int  de  = (lane < 8) ? lane : (lane - 24);   // element of v8f acc
  const int  nidx = own ? neigh[p0 + dp] : 0;

#pragma unroll
  for (int dim = 0; dim < 3; ++dim) {
    // A tile row for this lane, loaded straight from global in WMMA order
    // (contiguous 4KB tile; every 128B line consumed exactly once).
    const float* aRow = cd + ((size_t)dim * P + (size_t)(p0 + mn)) * NDESC;

    v8f acc = {};
    v16bf a0 = pack_half(aRow, 0 + cb);
    acc = __builtin_amdgcn_wmma_f32_16x16x32_bf16(
        false, a0, false, bmat[0], (short)0, acc, false, false);
    v16bf a1 = pack_half(aRow, 32 + cb);
    acc = __builtin_amdgcn_wmma_f32_16x16x32_bf16(
        false, a1, false, bmat[1], (short)0, acc, false, false);

    // Extract diagonal (the 16 wanted dot products) and scatter-add.
    float dot = acc[de];
    if (own) atomicAdd(&outF[(size_t)dim * NA + nidx], -dot);
  }
}

// Scalar tail for P % 16 != 0 (not hit for P=500000, kept for generality).
__global__ void pair_tail_kernel(const float* __restrict__ cd,
                                 const float* __restrict__ g,
                                 const int* __restrict__ central,
                                 const int* __restrict__ neigh,
                                 float* __restrict__ outF,
                                 int P, int NA, int pstart) {
  int np = P - pstart;
  int t = blockIdx.x * blockDim.x + threadIdx.x;
  if (t >= np * 3) return;
  int p = pstart + (t % np);
  int dim = t / np;
  const float* a = cd + ((size_t)dim * P + (size_t)p) * NDESC;
  const float* gr = g + (size_t)central[p] * NDESC;
  float dot = 0.0f;
  for (int k = 0; k < NDESC; ++k) dot += a[k] * gr[k];
  atomicAdd(&outF[(size_t)dim * NA + neigh[p]], -dot);
}

extern "C" void kernel_launch(void* const* d_in, const int* in_sizes, int n_in,
                              void* d_out, int out_size, void* d_ws,
                              size_t ws_size, hipStream_t stream) {
  const float* coeffs  = (const float*)d_in[0];  // [1, NA, 64]
  const float* cd      = (const float*)d_in[1];  // [1, 3, P, 64]
  const float* w1      = (const float*)d_in[2];  // [64]
  const float* b1      = (const float*)d_in[3];  // [64]
  const float* wl      = (const float*)d_in[4];  // [64]
  const float* bl      = (const float*)d_in[5];  // [1]
  const int*   central = (const int*)d_in[6];    // [P]
  const int*   neigh   = (const int*)d_in[7];    // [P]

  const int NA = in_sizes[0] / NDESC;            // 10000
  const int P  = in_sizes[6];                    // 500000
  const int total = NA * NDESC;

  float* out  = (float*)d_out;                   // [0]=energy, [1..]=forces
  float* outF = out + 1;                         // [3, NA]
  float* g    = (float*)d_ws;                    // [NA, 64] f32

  // 1) init output (energy starts at b_last, forces at 0)
  init_out_kernel<<<(out_size + 255) / 256, 256, 0, stream>>>(out, bl,
                                                              out_size);
  // 2) per-atom g-table + energy
  atom_kernel<<<(total + 255) / 256, 256, 0, stream>>>(
      coeffs, w1, b1, wl, g, out, total, 1.0f / (float)NA);
  // 3) per-pair WMMA streaming kernel
  const int nTiles = P / 16;
  if (nTiles > 0) {
    pair_wmma_kernel<<<(nTiles + 3) / 4, 128, 0, stream>>>(
        cd, g, central, neigh, outF, P, NA, nTiles);
  }
  const int rem = P - nTiles * 16;
  if (rem > 0) {
    pair_tail_kernel<<<(rem * 3 + 255) / 256, 256, 0, stream>>>(
        cd, g, central, neigh, outF, P, NA, nTiles * 16);
  }
}
